// Label2Text_73967926772198
// MI455X (gfx1250) — matrix operand order repaired and checked by reading, not compiled, not run on hardware
//
#include <hip/hip_runtime.h>
#include <hip/hip_bf16.h>
#include <math.h>

// ---------------- problem constants ----------------
constexpr int B = 16, S = 4096, L = 512, E = 1024, H = 1024;

typedef __bf16 bf16_t;
typedef __attribute__((ext_vector_type(16))) __bf16 bf16x16;
typedef __attribute__((ext_vector_type(8)))  float  f32x8;

// ---------------- workspace layout (bytes) ----------------
constexpr size_t XN_OFF   = 0;                                   // bf16 [B][S][H]
constexpr size_t XNT_OFF  = XN_OFF   + (size_t)B*S*H*2;          // bf16 [B][H][S]
constexpr size_t LAB_OFF  = XNT_OFF  + (size_t)B*S*H*2;          // f32  [B][L][H]
constexpr size_t LABN_OFF = LAB_OFF  + (size_t)B*L*H*4;          // bf16 [B][L][H]
constexpr size_t G_OFF    = LABN_OFF + (size_t)B*L*H*2;          // f32  [B][S][L]
constexpr size_t CMAX_OFF = G_OFF    + (size_t)B*S*L*4;          // f32  [B][L]
constexpr size_t CSUM_OFF = CMAX_OFF + (size_t)B*L*4;            // f32  [B][L]
constexpr size_t PT_OFF   = CSUM_OFF + (size_t)B*L*4;            // bf16 [B][L][S]

// ---------------- WMMA helpers ----------------
union FragCast {
  bf16x16 v;
  uint4   q[2];
};

__device__ __forceinline__ f32x8 wmma_bf16(bf16x16 a, bf16x16 b, f32x8 c) {
  // v_wmma_f32_16x16x32_bf16
  return __builtin_amdgcn_wmma_f32_16x16x32_bf16(
      /*neg_a=*/false, a, /*neg_b=*/false, b,
      /*c_mod=*/(short)0, c, /*reuse_a=*/false, /*reuse_b=*/false);
}

// A-matrix 16x32 bf16 fragment: lane(half,m) holds
// K = 16*half + {0..15}  (two contiguous 16B chunks)
__device__ __forceinline__ bf16x16 load_a_bf16(const bf16_t* rowk0, int half) {
  FragCast f;
  f.q[0] = *(const uint4*)(rowk0 + 16*half);
  f.q[1] = *(const uint4*)(rowk0 + 16*half + 8);
  return f.v;
}

// B-matrix 32x16 bf16 fragment: lane(half,n) holds K = 16*half + {0..15}
// (one contiguous 32B chunk of row n of the N-major source)
__device__ __forceinline__ bf16x16 load_b_bf16(const bf16_t* rowk0, int half) {
  FragCast f;
  const uint4* p = (const uint4*)(rowk0 + 16*half);
  f.q[0] = p[0];
  f.q[1] = p[1];
  return f.v;
}

// Same fragments but sourced from f32 rows (convert to bf16 on the fly)
__device__ __forceinline__ bf16x16 load_a_f32(const float* rowk0, int half) {
  const float* p0 = rowk0 + 16*half;
  const float* p1 = rowk0 + 16*half + 8;
  bf16x16 r;
#pragma unroll
  for (int i = 0; i < 8; ++i) {
    r[i]     = (bf16_t)p0[i];
    r[i + 8] = (bf16_t)p1[i];
  }
  return r;
}

__device__ __forceinline__ bf16x16 load_b_f32(const float* rowk0, int half) {
  const float* p = rowk0 + 16*half;
  bf16x16 r;
#pragma unroll
  for (int i = 0; i < 16; ++i) r[i] = (bf16_t)p[i];
  return r;
}

// ---------------- K1/K4: row L2-normalize (1024 cols) f32 -> bf16 ----------------
__global__ void k_rownorm(const float* __restrict__ in, bf16_t* __restrict__ out) {
  const int NC = 1024;
  const int row = blockIdx.x;
  const int t = threadIdx.x;  // 256 threads
  const float* rp = in + (size_t)row * NC;
  float v0 = rp[t], v1 = rp[t + 256], v2 = rp[t + 512], v3 = rp[t + 768];
  float ss = v0*v0 + v1*v1 + v2*v2 + v3*v3;
  __shared__ float red[256];
  red[t] = ss;
  __syncthreads();
  for (int off = 128; off; off >>= 1) {
    if (t < off) red[t] += red[t + off];
    __syncthreads();
  }
  float inv = 1.0f / fmaxf(sqrtf(red[0]), 1e-12f);
  bf16_t* op = out + (size_t)row * NC;
  op[t]       = (bf16_t)(v0 * inv);
  op[t + 256] = (bf16_t)(v1 * inv);
  op[t + 512] = (bf16_t)(v2 * inv);
  op[t + 768] = (bf16_t)(v3 * inv);
}

// ---------------- K2: batched tiled transpose bf16 [b][rows][cols] -> [b][cols][rows] ----------------
__global__ void k_transpose64(const bf16_t* __restrict__ src, bf16_t* __restrict__ dst,
                              int rows, int cols) {
  const int tilesR = rows >> 6, tilesC = cols >> 6;
  int id = blockIdx.x;
  int b  = id / (tilesR * tilesC);
  int r2 = id % (tilesR * tilesC);
  int tr = r2 / tilesC, tc = r2 % tilesC;
  size_t sbase = ((size_t)b * rows + tr * 64) * cols + (size_t)tc * 64;
  size_t dbase = ((size_t)b * cols + tc * 64) * rows + (size_t)tr * 64;
  __shared__ bf16_t tile[64][65];
  int t = threadIdx.x;  // 256
#pragma unroll
  for (int it = 0; it < 16; ++it) {
    int idx = it * 256 + t;
    int rr = idx >> 6, cc = idx & 63;
    tile[rr][cc] = src[sbase + (size_t)rr * cols + cc];
  }
  __syncthreads();
#pragma unroll
  for (int it = 0; it < 16; ++it) {
    int idx = it * 256 + t;
    int rr = idx >> 6, cc = idx & 63;
    dst[dbase + (size_t)rr * rows + cc] = tile[cc][rr];
  }
}

// ---------------- K3: projection GEMM  lab[b,l,h] = sum_e label[b,l,e]*W[h,e] ----------------
// wave tile 32x32, 4 waves per block, K = E
__global__ void k_gemm_proj(const float* __restrict__ label, const float* __restrict__ W,
                            float* __restrict__ LAB) {
  const int TM = L / 32, TN = H / 32;  // 16 x 32
  int wid  = blockIdx.x * 4 + (threadIdx.x >> 5);
  int lane = threadIdx.x & 31;
  int b  = wid / (TM * TN);
  int r  = wid % (TM * TN);
  int m0 = (r / TN) * 32, n0 = (r % TN) * 32;
  int half = lane >> 4, q = lane & 15;

  const float* arow0 = label + ((size_t)b * L + m0 + q) * E;
  const float* arow1 = arow0 + (size_t)16 * E;
  const float* brow0 = W + (size_t)(n0 + q) * E;
  const float* brow1 = brow0 + (size_t)16 * E;

  f32x8 c00 = {0,0,0,0,0,0,0,0}, c01 = c00, c10 = c00, c11 = c00;
  for (int k0 = 0; k0 < E; k0 += 32) {
    bf16x16 a0 = load_a_f32(arow0 + k0, half);
    bf16x16 a1 = load_a_f32(arow1 + k0, half);
    bf16x16 b0 = load_b_f32(brow0 + k0, half);
    bf16x16 b1 = load_b_f32(brow1 + k0, half);
    c00 = wmma_bf16(a0, b0, c00);
    c01 = wmma_bf16(a0, b1, c01);
    c10 = wmma_bf16(a1, b0, c10);
    c11 = wmma_bf16(a1, b1, c11);
  }
  float* Cb = LAB + (size_t)b * L * H;
#pragma unroll
  for (int r8 = 0; r8 < 8; ++r8) {
    int Mlo = m0 + 8 * half + r8;
    int Mhi = Mlo + 16;
    Cb[(size_t)Mlo * H + n0 + q]      = c00[r8];
    Cb[(size_t)Mlo * H + n0 + 16 + q] = c01[r8];
    Cb[(size_t)Mhi * H + n0 + q]      = c10[r8];
    Cb[(size_t)Mhi * H + n0 + 16 + q] = c11[r8];
  }
}

// ---------------- K5: score GEMM + mask  G[b,s,l] = x_n[b,s,:].lab_n[b,l,:] ----------------
__global__ void k_gemm_scores(const bf16_t* __restrict__ XN, const bf16_t* __restrict__ LABN,
                              const int* __restrict__ mask, float* __restrict__ G) {
  const int TM = S / 32, TN = L / 32;  // 128 x 16
  int wid  = blockIdx.x * 4 + (threadIdx.x >> 5);
  int lane = threadIdx.x & 31;
  int b  = wid / (TM * TN);
  int r  = wid % (TM * TN);
  int m0 = (r / TN) * 32, n0 = (r % TN) * 32;
  int half = lane >> 4, q = lane & 15;

  const bf16_t* arow0 = XN + ((size_t)b * S + m0 + q) * H;
  const bf16_t* arow1 = arow0 + (size_t)16 * H;
  const bf16_t* brow0 = LABN + ((size_t)b * L + n0 + q) * H;
  const bf16_t* brow1 = brow0 + (size_t)16 * H;

  f32x8 c00 = {0,0,0,0,0,0,0,0}, c01 = c00, c10 = c00, c11 = c00;
  for (int k0 = 0; k0 < H; k0 += 32) {
    if (k0 + 256 < H) {
      __builtin_prefetch(arow0 + k0 + 256);
      __builtin_prefetch(brow0 + k0 + 256);
    }
    bf16x16 a0 = load_a_bf16(arow0 + k0, half);
    bf16x16 a1 = load_a_bf16(arow1 + k0, half);
    bf16x16 b0 = load_b_bf16(brow0 + k0, half);
    bf16x16 b1 = load_b_bf16(brow1 + k0, half);
    c00 = wmma_bf16(a0, b0, c00);
    c01 = wmma_bf16(a0, b1, c01);
    c10 = wmma_bf16(a1, b0, c10);
    c11 = wmma_bf16(a1, b1, c11);
  }
  const int* mrow = mask + (size_t)b * S;
  float* Gb = G + (size_t)b * S * L;
  const float NEG_INF = -__builtin_inff();
#pragma unroll
  for (int r8 = 0; r8 < 8; ++r8) {
    int Mlo = m0 + 8 * half + r8;
    int Mhi = Mlo + 16;
    bool alo = mrow[Mlo] != 0;
    bool ahi = mrow[Mhi] != 0;
    Gb[(size_t)Mlo * L + n0 + q]      = alo ? c00[r8] : NEG_INF;
    Gb[(size_t)Mlo * L + n0 + 16 + q] = alo ? c01[r8] : NEG_INF;
    Gb[(size_t)Mhi * L + n0 + q]      = ahi ? c10[r8] : NEG_INF;
    Gb[(size_t)Mhi * L + n0 + 16 + q] = ahi ? c11[r8] : NEG_INF;
  }
}

// ---------------- K6: per-(b,l) max & sum(exp) over S ----------------
__global__ void k_colstats(const float* __restrict__ G, float* __restrict__ cmax,
                           float* __restrict__ csum) {
  int tx = threadIdx.x;            // 64 : l within chunk
  int ty = threadIdx.y;            // 16 : s strider
  int b  = blockIdx.y;
  int l  = blockIdx.x * 64 + tx;
  __shared__ float red[16][64];

  float m = -__builtin_inff();
  for (int s = ty; s < S; s += 16)
    m = fmaxf(m, G[((size_t)b * S + s) * L + l]);
  red[ty][tx] = m;
  __syncthreads();
  if (ty == 0) {
#pragma unroll
    for (int j = 1; j < 16; ++j) m = fmaxf(m, red[j][tx]);
    red[0][tx] = m;
  }
  __syncthreads();
  m = red[0][tx];
  __syncthreads();

  float sum = 0.0f;
  for (int s = ty; s < S; s += 16)
    sum += expf(G[((size_t)b * S + s) * L + l] - m);
  red[ty][tx] = sum;
  __syncthreads();
  if (ty == 0) {
#pragma unroll
    for (int j = 1; j < 16; ++j) sum += red[j][tx];
    cmax[(size_t)b * L + l] = m;
    csum[(size_t)b * L + l] = sum;
  }
}

// ---------------- K7: softmax -> d_out(f32 [B,S,L]) + transposed bf16 Pt [B,L,S] ----------------
__global__ void k_softmax_write(const float* __restrict__ G, const float* __restrict__ cmax,
                                const float* __restrict__ csum, float* __restrict__ soft,
                                bf16_t* __restrict__ Pt) {
  const int tS = S / 64, tL = L / 64;  // 64 x 8 tiles per batch
  int id = blockIdx.x;
  int b  = id / (tS * tL);
  int r  = id % (tS * tL);
  int s0 = (r / tL) * 64, l0 = (r % tL) * 64;
  int t = threadIdx.x;  // 256
  __shared__ bf16_t tile[64][65];
  __shared__ float mx[64], sm[64];
  if (t < 64) {
    mx[t] = cmax[(size_t)b * L + l0 + t];
    sm[t] = csum[(size_t)b * L + l0 + t];
  }
  __syncthreads();
#pragma unroll
  for (int it = 0; it < 16; ++it) {
    int idx = it * 256 + t;
    int sr = idx >> 6, lc = idx & 63;
    size_t gi = ((size_t)b * S + s0 + sr) * L + l0 + lc;
    float p = expf(G[gi] - mx[lc]) / sm[lc];  // masked rows: exp(-inf - m) == 0
    soft[gi] = p;
    tile[sr][lc] = (bf16_t)p;
  }
  __syncthreads();
#pragma unroll
  for (int it = 0; it < 16; ++it) {
    int idx = it * 256 + t;
    int lr = idx >> 6, sc = idx & 63;
    Pt[((size_t)b * L + l0 + lr) * S + s0 + sc] = tile[sc][lr];
  }
}

// ---------------- K8: output GEMM  out[b,l,h] = sum_s P[b,s,l]*x_n[b,s,h] ----------------
// A = Pt [B][L][S] (K = s contiguous), B = XNT [B][H][S] (K = s contiguous)
__global__ void k_gemm_out(const bf16_t* __restrict__ PT, const bf16_t* __restrict__ XNT,
                           float* __restrict__ out) {
  const int TM = L / 32, TN = H / 32;  // 16 x 32
  int wid  = blockIdx.x * 4 + (threadIdx.x >> 5);
  int lane = threadIdx.x & 31;
  int b  = wid / (TM * TN);
  int r  = wid % (TM * TN);
  int m0 = (r / TN) * 32, n0 = (r % TN) * 32;
  int half = lane >> 4, q = lane & 15;

  const bf16_t* arow0 = PT + ((size_t)b * L + m0 + q) * S;
  const bf16_t* arow1 = arow0 + (size_t)16 * S;
  const bf16_t* brow0 = XNT + ((size_t)b * H + n0 + q) * S;
  const bf16_t* brow1 = brow0 + (size_t)16 * S;

  f32x8 c00 = {0,0,0,0,0,0,0,0}, c01 = c00, c10 = c00, c11 = c00;
  for (int k0 = 0; k0 < S; k0 += 32) {
    if (k0 + 256 < S) {
      __builtin_prefetch(arow0 + k0 + 256);
      __builtin_prefetch(brow0 + k0 + 256);
    }
    bf16x16 a0 = load_a_bf16(arow0 + k0, half);
    bf16x16 a1 = load_a_bf16(arow1 + k0, half);
    bf16x16 b0 = load_b_bf16(brow0 + k0, half);
    bf16x16 b1 = load_b_bf16(brow1 + k0, half);
    c00 = wmma_bf16(a0, b0, c00);
    c01 = wmma_bf16(a0, b1, c01);
    c10 = wmma_bf16(a1, b0, c10);
    c11 = wmma_bf16(a1, b1, c11);
  }
  float* Cb = out + (size_t)b * L * H;
#pragma unroll
  for (int r8 = 0; r8 < 8; ++r8) {
    int Mlo = m0 + 8 * half + r8;
    int Mhi = Mlo + 16;
    Cb[(size_t)Mlo * H + n0 + q]      = c00[r8];
    Cb[(size_t)Mlo * H + n0 + 16 + q] = c01[r8];
    Cb[(size_t)Mhi * H + n0 + q]      = c10[r8];
    Cb[(size_t)Mhi * H + n0 + 16 + q] = c11[r8];
  }
}

// ---------------- launcher ----------------
extern "C" void kernel_launch(void* const* d_in, const int* in_sizes, int n_in,
                              void* d_out, int out_size, void* d_ws, size_t ws_size,
                              hipStream_t stream) {
  (void)in_sizes; (void)n_in; (void)out_size; (void)ws_size;
  const float* input = (const float*)d_in[0];   // [B,S,H] f32
  const float* label = (const float*)d_in[1];   // [B,L,E] f32
  const float* Wp    = (const float*)d_in[2];   // [H,E]   f32
  const int*   mask  = (const int*)d_in[3];     // [B,S]   i32

  float* out  = (float*)d_out;                   // [B,L,H] f32
  float* soft = out + (size_t)B * L * H;         // [B,S,L] f32

  char* ws = (char*)d_ws;
  bf16_t* XN   = (bf16_t*)(ws + XN_OFF);
  bf16_t* XNT  = (bf16_t*)(ws + XNT_OFF);
  float*  LAB  = (float*) (ws + LAB_OFF);
  bf16_t* LABN = (bf16_t*)(ws + LABN_OFF);
  float*  G    = (float*) (ws + G_OFF);
  float*  CMAX = (float*) (ws + CMAX_OFF);
  float*  CSUM = (float*) (ws + CSUM_OFF);
  bf16_t* PT   = (bf16_t*)(ws + PT_OFF);

  // 1) x_n = l2norm(input) -> bf16
  k_rownorm<<<B * S, 256, 0, stream>>>(input, XN);
  // 2) x_n^T for output GEMM
  k_transpose64<<<B * (S / 64) * (H / 64), 256, 0, stream>>>(XN, XNT, S, H);
  // 3) lab = label @ W^T   (WMMA bf16)
  k_gemm_proj<<<(B * (L / 32) * (H / 32)) / 4, 128, 0, stream>>>(label, Wp, LAB);
  // 4) lab_n = l2norm(lab) -> bf16
  k_rownorm<<<B * L, 256, 0, stream>>>(LAB, LABN);
  // 5) G = x_n . lab_n^T + mask   (WMMA bf16)
  k_gemm_scores<<<(B * (S / 32) * (L / 32)) / 4, 128, 0, stream>>>(XN, LABN, mask, G);
  // 6) softmax stats over S
  k_colstats<<<dim3(L / 64, B), dim3(64, 16), 0, stream>>>(G, CMAX, CSUM);
  // 7) softmax -> d_out (f32) + P^T bf16
  k_softmax_write<<<B * (S / 64) * (L / 64), 256, 0, stream>>>(G, CMAX, CSUM, soft, PT);
  // 8) out = P^T @ x_n   (WMMA bf16)
  k_gemm_out<<<(B * (L / 32) * (H / 32)) / 4, 128, 0, stream>>>(PT, XNT, out);
}
